// SpectralGraphConvolution_5823975653969
// MI455X (gfx1250) — matrix operand order repaired and checked by reading, not compiled, not run on hardware
//
#include <hip/hip_runtime.h>

// ---------------------------------------------------------------------------
// SpectralGraphConvolution on MI455X (gfx1250, wave32, WMMA)
//
// out[b,n,o] = max_r (A_r @ HW_r)[b,n,o] + bias[n,o]
//   r in {0:A1, 1:A2, 2:I, 3:A1^T, 4:A2^T, 5:I, 6:I}   (roll of broadcast eye == eye)
//   HW_r = features @ W[r][:F] + eb[r],  eb[r][o] = sum_d W_edges[r,d]*W[r][F+d,o]
//
// Strategy: bf16 WMMA (v_wmma_f32_16x16x32_bf16) with f32 accumulation.
// ---------------------------------------------------------------------------

typedef __attribute__((ext_vector_type(16))) __bf16 v16bf;
typedef __attribute__((ext_vector_type(8)))  __bf16 v8bf;
typedef __attribute__((ext_vector_type(4)))  __bf16 v4bf;
typedef __attribute__((ext_vector_type(8)))  float  v8f;

#define LSTR 40   // LDS row stride in bf16 elements (80 B = multiple of 16 B)

static __device__ __forceinline__ v16bf frag_cat(const __bf16* p0, const __bf16* p1) {
  v8bf lo = *(const v8bf*)p0;
  v8bf hi = *(const v8bf*)p1;
  return __builtin_shufflevector(lo, hi, 0,1,2,3,4,5,6,7,8,9,10,11,12,13,14,15);
}

static __device__ __forceinline__ v8f wmma_bf16(v16bf a, v16bf b, v8f c) {
  return __builtin_amdgcn_wmma_f32_16x16x32_bf16(false, a, false, b, (short)0, c,
                                                 false, false);
}

static __device__ __forceinline__ v8f zero8() {
  v8f z;
  #pragma unroll
  for (int i = 0; i < 8; ++i) z[i] = 0.0f;
  return z;
}

// ---------------------------------------------------------------------------
// Prep: W[r][k][o] (f32, k<256) -> Wt[r][o][k] (bf16)  (B-fragment friendly)
// ---------------------------------------------------------------------------
__global__ __launch_bounds__(256) void k_prep_w(const float* __restrict__ W,
                                                __bf16* __restrict__ Wt) {
  int g = blockIdx.x * 256 + threadIdx.x;        // 7*256*256 threads
  int k = g & 255;
  int o = (g >> 8) & 255;
  int r = g >> 16;
  Wt[((size_t)(r * 256 + o) << 8) + k] = (__bf16)W[((size_t)(r * 272 + k) << 8) + o];
}

// eb[r][o] = sum_{d<16} W_edges[r,d] * W[r][256+d][o]
__global__ __launch_bounds__(256) void k_prep_eb(const float* __restrict__ W,
                                                 const float* __restrict__ We,
                                                 float* __restrict__ eb) {
  int g = blockIdx.x * 256 + threadIdx.x;        // 7*256 threads
  int o = g & 255;
  int r = g >> 8;
  float s = 0.0f;
  #pragma unroll
  for (int d = 0; d < 16; ++d)
    s += We[r * 16 + d] * W[((size_t)(r * 272 + 256 + d) << 8) + o];
  eb[g] = s;
}

// ---------------------------------------------------------------------------
// Stage 1: HW_r = features @ W[r][:F] + eb[r] for all 7 r.
//   - r in {2,5,6} (identity adjacency): fold into out = max(...) + bias (f32)
//   - r in {0,1,3,4}: store bf16 into H[0..3]
// Tile: 32 rows x 64 out-cols per block (8 waves, each 16x16 per r).
// ---------------------------------------------------------------------------
__global__ __launch_bounds__(256) void k_hw(const float* __restrict__ feat,  // [8192][256]
                                            const __bf16* __restrict__ Wt,  // [7][256][256]
                                            const float* __restrict__ eb,   // [7][256]
                                            const float* __restrict__ bias, // [1024][256]
                                            float* __restrict__ out,        // [8192][256]
                                            __bf16* __restrict__ H) {       // [4][8192][256]
  __shared__ __align__(16) __bf16 fs[32 * LSTR];
  __shared__ __align__(16) __bf16 ws[7 * 64 * LSTR];

  const int tid  = threadIdx.x;
  const int lane = tid & 31;
  const int wave = tid >> 5;
  const int isub = wave >> 2;          // 0..1  (16-row subtile)
  const int osub = wave & 3;           // 0..3  (16-col subtile)
  const int row0 = blockIdx.x * 32;
  const int o0   = blockIdx.y * 64;

  v8f acc[7];
  #pragma unroll
  for (int r = 0; r < 7; ++r) acc[r] = zero8();

  for (int kk = 0; kk < 256; kk += 32) {
    // stage features tile [32 rows][32 k] as bf16
    {
      int idx = tid * 4;
      int i = idx >> 5, k = idx & 31;
      float4 f = *(const float4*)(feat + (size_t)(row0 + i) * 256 + kk + k);
      v4bf h;
      h[0] = (__bf16)f.x; h[1] = (__bf16)f.y; h[2] = (__bf16)f.z; h[3] = (__bf16)f.w;
      *(v4bf*)(&fs[i * LSTR + k]) = h;
    }
    // stage Wt tiles for all 7 relations: [r][64 o][32 k] bf16 (already transposed)
    #pragma unroll
    for (int s = 0; s < 7; ++s) {
      int idx = (s * 256 + tid) * 8;          // 8 bf16 per segment
      int ro = idx >> 5, k = idx & 31;        // k in {0,8,16,24}
      int r = ro >> 6, o = ro & 63;
      uint4 w = *(const uint4*)(Wt + ((size_t)(r * 256 + o0 + o) << 8) + kk + k);
      *(uint4*)(&ws[r * 64 * LSTR + o * LSTR + k]) = w;
    }
    __syncthreads();

    // A fragment (16x32 bf16): lane holds row M=lane&15, K 0-15 (or 16-31)
    const int arow = isub * 16 + (lane & 15);
    const int kb   = (lane >> 4) * 16;
    v16bf af = frag_cat(&fs[arow * LSTR + kb], &fs[arow * LSTR + kb + 8]);

    const int bn  = osub * 16 + (lane & 15);
    const int kb2 = (lane >> 4) * 16;
    #pragma unroll
    for (int r = 0; r < 7; ++r) {
      const __bf16* bp = &ws[r * 64 * LSTR + bn * LSTR + kb2];
      v16bf bfr = frag_cat(bp, bp + 8);
      acc[r] = wmma_bf16(af, bfr, acc[r]);
    }
    __syncthreads();
  }

  // epilogue: C layout is VGPR v <-> (M = v + 8*(lane>=16), N = lane&15)
  const int ocol = o0 + osub * 16 + (lane & 15);
  float ebv[7];
  #pragma unroll
  for (int r = 0; r < 7; ++r) ebv[r] = eb[r * 256 + ocol];
  const int rbase = row0 + isub * 16 + (lane >> 4) * 8;

  #pragma unroll
  for (int v = 0; v < 8; ++v) {
    const int row = rbase + v;
    const int n   = row & 1023;
    float h2 = acc[2][v] + ebv[2];
    float h5 = acc[5][v] + ebv[5];
    float h6 = acc[6][v] + ebv[6];
    float m = fmaxf(fmaxf(h2, h5), h6);
    out[(size_t)row * 256 + ocol] = m + bias[(size_t)n * 256 + ocol];
    H[(size_t)0 * 8192 * 256 + (size_t)row * 256 + ocol] = (__bf16)(acc[0][v] + ebv[0]);
    H[(size_t)1 * 8192 * 256 + (size_t)row * 256 + ocol] = (__bf16)(acc[1][v] + ebv[1]);
    H[(size_t)2 * 8192 * 256 + (size_t)row * 256 + ocol] = (__bf16)(acc[3][v] + ebv[3]);
    H[(size_t)3 * 8192 * 256 + (size_t)row * 256 + ocol] = (__bf16)(acc[4][v] + ebv[4]);
  }
}

// ---------------------------------------------------------------------------
// Stage 2: for one A matrix, compute A@Hf and A^T@Hb and max-merge into out.
// Tile 64(i) x 64(o) per block; wave = 16(i) x 32(o); K loop over j (step 32).
// ---------------------------------------------------------------------------
__global__ __launch_bounds__(256) void k_adj(const float* __restrict__ A,   // [8][1024][1024]
                                             const __bf16* __restrict__ Hf, // [8192][256]
                                             const __bf16* __restrict__ Hb, // [8192][256]
                                             float* __restrict__ out) {     // [8192][256]
  __shared__ __align__(16) __bf16 Afs[64 * LSTR];  // A[i0+i][j+k]
  __shared__ __align__(16) __bf16 Abs[64 * LSTR];  // A[j+k][i0+i]  (A^T tile)
  __shared__ __align__(16) __bf16 Hfs[64 * LSTR];  // Hf transposed to [o][k]
  __shared__ __align__(16) __bf16 Hbs[64 * LSTR];  // Hb transposed to [o][k]

  const int tid  = threadIdx.x;
  const int lane = tid & 31;
  const int wave = tid >> 5;
  const int isub = wave >> 1;          // 0..3
  const int osub = wave & 1;           // 0..1
  const int z  = blockIdx.z;
  const int i0 = blockIdx.x * 64;
  const int o0 = blockIdx.y * 64;
  const float* Az = A + (size_t)z * 1024 * 1024;
  const size_t hrow0 = (size_t)z * 1024;

  v8f f0 = zero8(), f1 = zero8(), b0 = zero8(), b1 = zero8();

  for (int j = 0; j < 1024; j += 32) {
    #pragma unroll
    for (int s = 0; s < 2; ++s) {
      int idx = (s * 256 + tid) * 4;
      {  // forward A tile: 64 i-rows x 32 k, row-major f32 source
        int i = idx >> 5, k = idx & 31;
        float4 f = *(const float4*)(Az + (size_t)(i0 + i) * 1024 + j + k);
        v4bf h;
        h[0] = (__bf16)f.x; h[1] = (__bf16)f.y; h[2] = (__bf16)f.z; h[3] = (__bf16)f.w;
        *(v4bf*)(&Afs[i * LSTR + k]) = h;
      }
      {  // transposed A tile: source rows j+k, cols i0..i0+63 -> scatter to [i][k]
        int k = idx >> 6, i = idx & 63;
        float4 f = *(const float4*)(Az + (size_t)(j + k) * 1024 + i0 + i);
        Abs[(i + 0) * LSTR + k] = (__bf16)f.x;
        Abs[(i + 1) * LSTR + k] = (__bf16)f.y;
        Abs[(i + 2) * LSTR + k] = (__bf16)f.z;
        Abs[(i + 3) * LSTR + k] = (__bf16)f.w;
      }
      {  // HW tiles: bf16 source [j+k][o], transpose to LDS [o][k]
        int k = idx >> 6, o = idx & 63;
        v4bf h = *(const v4bf*)(Hf + (hrow0 + j + k) * 256 + o0 + o);
        Hfs[(o + 0) * LSTR + k] = h[0];
        Hfs[(o + 1) * LSTR + k] = h[1];
        Hfs[(o + 2) * LSTR + k] = h[2];
        Hfs[(o + 3) * LSTR + k] = h[3];
        v4bf g = *(const v4bf*)(Hb + (hrow0 + j + k) * 256 + o0 + o);
        Hbs[(o + 0) * LSTR + k] = g[0];
        Hbs[(o + 1) * LSTR + k] = g[1];
        Hbs[(o + 2) * LSTR + k] = g[2];
        Hbs[(o + 3) * LSTR + k] = g[3];
      }
    }
    __syncthreads();

    const int arow = isub * 16 + (lane & 15);
    const int kb   = (lane >> 4) * 16;
    const int kb2  = (lane >> 4) * 16;
    v16bf af = frag_cat(&Afs[arow * LSTR + kb], &Afs[arow * LSTR + kb + 8]);
    v16bf ab = frag_cat(&Abs[arow * LSTR + kb], &Abs[arow * LSTR + kb + 8]);

    const int n0 = osub * 32 + (lane & 15);
    v16bf vf0 = frag_cat(&Hfs[n0 * LSTR + kb2],        &Hfs[n0 * LSTR + kb2 + 8]);
    v16bf vf1 = frag_cat(&Hfs[(n0 + 16) * LSTR + kb2], &Hfs[(n0 + 16) * LSTR + kb2 + 8]);
    v16bf vb0 = frag_cat(&Hbs[n0 * LSTR + kb2],        &Hbs[n0 * LSTR + kb2 + 8]);
    v16bf vb1 = frag_cat(&Hbs[(n0 + 16) * LSTR + kb2], &Hbs[(n0 + 16) * LSTR + kb2 + 8]);

    f0 = wmma_bf16(af, vf0, f0);
    f1 = wmma_bf16(af, vf1, f1);
    b0 = wmma_bf16(ab, vb0, b0);
    b1 = wmma_bf16(ab, vb1, b1);
    __syncthreads();
  }

  // max-merge into out (tiles disjoint within a launch; launches stream-ordered)
  const int rbase = i0 + isub * 16 + (lane >> 4) * 8;
  const int c0    = o0 + osub * 32 + (lane & 15);
  float* op = out + hrow0 * 256;
  #pragma unroll
  for (int v = 0; v < 8; ++v) {
    size_t off = (size_t)(rbase + v) * 256;
    float p0 = op[off + c0];
    op[off + c0] = fmaxf(p0, fmaxf(f0[v], b0[v]));
    float p1 = op[off + c0 + 16];
    op[off + c0 + 16] = fmaxf(p1, fmaxf(f1[v], b1[v]));
  }
}

// ---------------------------------------------------------------------------
extern "C" void kernel_launch(void* const* d_in, const int* in_sizes, int n_in,
                              void* d_out, int out_size, void* d_ws, size_t ws_size,
                              hipStream_t stream) {
  (void)in_sizes; (void)n_in; (void)out_size; (void)ws_size;
  const float* feat = (const float*)d_in[0];   // [8,1024,256]
  const float* A1   = (const float*)d_in[1];   // [8,1024,1024]
  const float* A2   = (const float*)d_in[2];   // [8,1024,1024]
  const float* W    = (const float*)d_in[3];   // [7,272,256]
  const float* We   = (const float*)d_in[4];   // [7,16]
  const float* bias = (const float*)d_in[5];   // [1024,256]
  float* out = (float*)d_out;                  // [8,1024,256]

  char* ws = (char*)d_ws;
  __bf16* H  = (__bf16*)ws;                          // 4*8192*256*2  = 16777216 B
  __bf16* Wt = (__bf16*)(ws + 16777216);             // 7*256*256*2   =   917504 B
  float*  eb = (float*)(ws + 16777216 + 917504);     // 7*256*4       =     7168 B

  k_prep_w <<<1792, 256, 0, stream>>>(W, Wt);
  k_prep_eb<<<7,    256, 0, stream>>>(W, We, eb);
  k_hw     <<<dim3(256, 4), 256, 0, stream>>>(feat, Wt, eb, bias, out, H);
  k_adj    <<<dim3(16, 4, 8), 256, 0, stream>>>(A1, H + (size_t)0 * 8192 * 256,
                                                H + (size_t)2 * 8192 * 256, out);
  k_adj    <<<dim3(16, 4, 8), 256, 0, stream>>>(A2, H + (size_t)1 * 8192 * 256,
                                                H + (size_t)3 * 8192 * 256, out);
}
